// Attn_LSTM_86912958202723
// MI455X (gfx1250) — matrix operand order repaired and checked by reading, not compiled, not run on hardware
//
#include <hip/hip_runtime.h>
#include <hip/hip_bf16.h>

#define HDIM 64
#define CDIM 8
#define LDIM 96
#define TDIM 24
#define BDIM 4096
#define BT   16    // batch tile per workgroup (WMMA M)
#define NTH  128   // 4 waves of 32

typedef __bf16 bf16_t;
typedef bf16_t v16bf __attribute__((ext_vector_type(16)));
typedef bf16_t v8bf  __attribute__((ext_vector_type(8)));
typedef float  v8f   __attribute__((ext_vector_type(8)));

// fast transcendentals: v_rcp_f32 + v_exp_f32, accurate far beyond bf16 data path
__device__ __forceinline__ float fast_sigmoid(float x) {
  return __builtin_amdgcn_rcpf(1.0f + __expf(-x));
}
__device__ __forceinline__ float fast_tanh(float x) {
  // tanh(x) = 1 - 2/(1+e^{2x}); saturates correctly at +/-1
  return 1.0f - 2.0f * __builtin_amdgcn_rcpf(1.0f + __expf(2.0f * x));
}

__device__ __forceinline__ v8f wmma_bf16(v16bf a, v16bf b, v8f c) {
  return __builtin_amdgcn_wmma_f32_16x16x32_bf16(false, a, false, b, (short)0, c, false, false);
}

// A fragment: 16xK row-major source (LDS), chunk base k0 (multiple of 32).
// lane<16: elems 0..15 = K k0+0..15
// lane>=16: elems 0..15 = K k0+16..31
__device__ __forceinline__ v16bf a_frag(const bf16_t* s, int ldk, int k0) {
  int lane = threadIdx.x & 31;
  int m = lane & 15;
  int kb = k0 + ((lane & 16) ? 16 : 0);
  const v8bf* p0 = (const v8bf*)(s + m * ldk + kb);
  const v8bf* p1 = (const v8bf*)(s + m * ldk + kb + 8);
  union { v16bf v; v8bf h[2]; } u;
  u.h[0] = *p0;
  u.h[1] = *p1;
  return u.v;
}

// B fragment: weight stored [N][K] row-major (out_features x in_features), bf16.
// lane n = N tile row; lane<16 covers K k0..k0+15, lane>=16 covers K k0+16..k0+31.
__device__ __forceinline__ v16bf b_frag(const bf16_t* w, int K, int n0, int k0) {
  int lane = threadIdx.x & 31;
  int n = n0 + (lane & 15);
  int kb = k0 + ((lane & 16) ? 16 : 0);
  return *(const v16bf*)(w + n * K + kb);
}

// D scatter to f32 LDS [16][ldn]: element r -> M = r + (lane>=16 ? 8:0), N = n0 + (lane&15)
__device__ __forceinline__ void scatter_d(float* s, int ldn, int n0, v8f d) {
  int lane = threadIdx.x & 31;
  int n = n0 + (lane & 15);
  int mb = (lane & 16) ? 8 : 0;
#pragma unroll
  for (int r = 0; r < 8; ++r) s[(mb + r) * ldn + n] = d[r];
}

__device__ __forceinline__ v8f bias_splat(const float* b, int n0) {
  int lane = threadIdx.x & 31;
  float bv = b[n0 + (lane & 15)];
  v8f acc;
#pragma unroll
  for (int r = 0; r < 8; ++r) acc[r] = bv;
  return acc;
}

// ---------------- prep kernels: f32 -> bf16 weight conversion with K padding ---------
__global__ void cvt_pad_kernel(const float* __restrict__ src, bf16_t* __restrict__ dst,
                               int N, int Ks, int Kd) {
  int i = blockIdx.x * blockDim.x + threadIdx.x;
  if (i >= N * Kd) return;
  int n = i / Kd, k = i - n * Kd;
  dst[i] = (k < Ks) ? (bf16_t)src[n * Ks + k] : (bf16_t)0.0f;
}

__global__ void bias_add_kernel(const float* __restrict__ a, const float* __restrict__ b,
                                float* __restrict__ d, int n) {
  int i = blockIdx.x * blockDim.x + threadIdx.x;
  if (i < n) d[i] = a[i] + b[i];
}

// ---------------- main fused kernel: one batch tile of 16 rows per workgroup ---------
__global__ __launch_bounds__(NTH)
void attn_lstm_main(const float* __restrict__ x_enc,
                    const bf16_t* __restrict__ wEncIh,   // 256 x 32 (K padded from 8)
                    const bf16_t* __restrict__ wEncHh,   // 256 x 64
                    const float*  __restrict__ bEnc,     // 256 (bih+bhh)
                    const bf16_t* __restrict__ wEmb,     // 64 x 32 (K padded from 8)
                    const float*  __restrict__ bEmb,     // 64
                    const bf16_t* __restrict__ wAttn,    // 96 x 128
                    const float*  __restrict__ bAttn,    // 96
                    const bf16_t* __restrict__ wComb,    // 64 x 128
                    const float*  __restrict__ bComb,    // 64
                    const bf16_t* __restrict__ wDecIh,   // 256 x 64
                    const bf16_t* __restrict__ wDecHh,   // 256 x 64
                    const float*  __restrict__ bDec,     // 256 (bih+bhh)
                    const float*  __restrict__ wOut,     // 8 x 64
                    const float*  __restrict__ bOut,     // 8
                    float* __restrict__ out)             // B x T x C
{
  __shared__ bf16_t sA[BT * 32];        // staged input tile (K padded to 32, zeros beyond C)
  __shared__ bf16_t sCat[BT * 128];     // [emb | h] or [emb | ctx] concat, bf16 A operand
  __shared__ bf16_t sB[BT * 64];        // comb result, bf16 A operand for decoder LSTM
  __shared__ bf16_t hbf[BT * 64];       // hidden state, bf16 A operand
  __shared__ float  hf[BT * 64];        // hidden state f32 (out projection)
  __shared__ float  cf[BT * 64];        // cell state f32
  __shared__ float  sG[BT * 256];       // gate / logit staging (f32)
  __shared__ float  sSeq[BT * CDIM];    // seq_last
  __shared__ float  sP[BT * CDIM];      // decoder input (pred feedback)
  extern __shared__ bf16_t encL[];      // LDIM x BT x HDIM bf16 = 192KB (attention source)

  const int tid = threadIdx.x;
  const int wv  = tid >> 5;
  const int b0  = blockIdx.x * BT;

  for (int i = tid; i < BT * 64; i += NTH) { hbf[i] = (bf16_t)0.0f; hf[i] = 0.0f; cf[i] = 0.0f; }
  for (int i = tid; i < BT * 32; i += NTH) sA[i] = (bf16_t)0.0f;
  if (tid < BT * CDIM) {
    int m = tid >> 3, c = tid & 7;
    sSeq[tid] = x_enc[(size_t)(b0 + m) * LDIM * CDIM + (LDIM - 1) * CDIM + c];
    sP[tid] = 0.0f;   // dec_in0 = x[:, -1, :] - seq_last == 0
  }
  __syncthreads();

  // ================= encoder: 96 LSTM steps =================
  for (int l = 0; l < LDIM; ++l) {
    if (tid < BT * CDIM) {
      int m = tid >> 3, c = tid & 7;
      float xv = x_enc[(size_t)(b0 + m) * LDIM * CDIM + l * CDIM + c] - sSeq[tid];
      sA[m * 32 + c] = (bf16_t)xv;
    }
    if ((l + 1) < LDIM && tid < BT)
      __builtin_prefetch(&x_enc[(size_t)(b0 + tid) * LDIM * CDIM + (l + 1) * CDIM], 0, 1);
    __syncthreads();

    // gates = x@Wih^T + h@Whh^T + b : 16x256, wave wv owns gate wv (4 N-tiles)
#pragma unroll
    for (int j = 0; j < 4; ++j) {
      int n0 = (wv * 4 + j) * 16;
      v8f acc = bias_splat(bEnc, n0);
      acc = wmma_bf16(a_frag(sA, 32, 0),  b_frag(wEncIh, 32, n0, 0),  acc);
      acc = wmma_bf16(a_frag(hbf, 64, 0), b_frag(wEncHh, 64, n0, 0),  acc);
      acc = wmma_bf16(a_frag(hbf, 64, 32),b_frag(wEncHh, 64, n0, 32), acc);
      scatter_d(sG, 256, n0, acc);
    }
    __syncthreads();

    for (int i = tid; i < BT * 64; i += NTH) {
      int m = i >> 6, h = i & 63;
      const float* g = &sG[m * 256];
      float ig = fast_sigmoid(g[h]);
      float fg = fast_sigmoid(g[64 + h]);
      float gg = fast_tanh(g[128 + h]);
      float og = fast_sigmoid(g[192 + h]);
      float cn = fg * cf[i] + ig * gg;
      float hn = og * fast_tanh(cn);
      cf[i] = cn; hf[i] = hn; hbf[i] = (bf16_t)hn;
      encL[(l * BT + m) * 64 + h] = (bf16_t)hn;   // enc_out tile stays in LDS
    }
    __syncthreads();
  }

  // ================= decoder: 24 attention-LSTM steps =================
  for (int t = 0; t < TDIM; ++t) {
    if (tid < BT * CDIM) {
      int m = tid >> 3, c = tid & 7;
      sA[m * 32 + c] = (bf16_t)sP[tid];
    }
    for (int i = tid; i < BT * 64; i += NTH) {       // cat second half = h (pre-update)
      int m = i >> 6, h = i & 63;
      sCat[m * 128 + 64 + h] = hbf[i];
    }
    __syncthreads();

    // emb = relu(inp @ embW^T + b): 16x64, wave wv owns N-tile wv
    {
      int n0 = wv * 16;
      v8f acc = bias_splat(bEmb, n0);
      acc = wmma_bf16(a_frag(sA, 32, 0), b_frag(wEmb, 32, n0, 0), acc);
      int lane = tid & 31;
      int n = n0 + (lane & 15);
      int mb = (lane & 16) ? 8 : 0;
#pragma unroll
      for (int r = 0; r < 8; ++r) {
        float v = acc[r] > 0.0f ? acc[r] : 0.0f;
        sCat[(mb + r) * 128 + n] = (bf16_t)v;        // cat first half = emb
      }
    }
    __syncthreads();

    // attention logits = cat(emb,h) @ attnW^T + b : 16x96, 6 N-tiles over 4 waves
    for (int nt = wv; nt < 6; nt += 4) {
      int n0 = nt * 16;
      v8f acc = bias_splat(bAttn, n0);
#pragma unroll
      for (int k0 = 0; k0 < 128; k0 += 32)
        acc = wmma_bf16(a_frag(sCat, 128, k0), b_frag(wAttn, 128, n0, k0), acc);
      scatter_d(sG, 256, n0, acc);
    }
    __syncthreads();

    // softmax over L per batch row: 8 threads/row (8 contiguous lanes of one wave),
    // width-8 shuffle reductions keep all 128 threads busy
    {
      int m = tid >> 3;     // 0..15
      int j = tid & 7;
      float* row = &sG[m * 256];
      float mx = -1e30f;
      for (int l2 = j; l2 < LDIM; l2 += 8) mx = fmaxf(mx, row[l2]);
      mx = fmaxf(mx, __shfl_xor(mx, 1, 8));
      mx = fmaxf(mx, __shfl_xor(mx, 2, 8));
      mx = fmaxf(mx, __shfl_xor(mx, 4, 8));
      float s = 0.0f;
      for (int l2 = j; l2 < LDIM; l2 += 8) { float e = __expf(row[l2] - mx); row[l2] = e; s += e; }
      s += __shfl_xor(s, 1, 8);
      s += __shfl_xor(s, 2, 8);
      s += __shfl_xor(s, 4, 8);
      float inv = __builtin_amdgcn_rcpf(s);
      for (int l2 = j; l2 < LDIM; l2 += 8) row[l2] *= inv;
    }
    __syncthreads();

    // ctx[m][h] = sum_l aw[m][l] * enc_out[m][l][h]  (pure LDS traffic)
    for (int i = tid; i < BT * 64; i += NTH) {
      int m = i >> 6, h = i & 63;
      const float* aw = &sG[m * 256];
      float s = 0.0f;
      for (int l2 = 0; l2 < LDIM; ++l2) s += aw[l2] * (float)encL[(l2 * BT + m) * 64 + h];
      sCat[m * 128 + 64 + h] = (bf16_t)s;            // cat second half = ctx
    }
    __syncthreads();

    // comb = cat(emb,ctx) @ combW^T + b : 16x64
    {
      int n0 = wv * 16;
      v8f acc = bias_splat(bComb, n0);
#pragma unroll
      for (int k0 = 0; k0 < 128; k0 += 32)
        acc = wmma_bf16(a_frag(sCat, 128, k0), b_frag(wComb, 128, n0, k0), acc);
      int lane = tid & 31;
      int n = n0 + (lane & 15);
      int mb = (lane & 16) ? 8 : 0;
#pragma unroll
      for (int r = 0; r < 8; ++r) sB[(mb + r) * 64 + n] = (bf16_t)acc[r];
    }
    __syncthreads();

    // decoder LSTM gates = comb@Wih^T + h@Whh^T + b : 16x256
#pragma unroll
    for (int j = 0; j < 4; ++j) {
      int n0 = wv * 64 + j * 16;
      v8f acc = bias_splat(bDec, n0);
      acc = wmma_bf16(a_frag(sB, 64, 0),   b_frag(wDecIh, 64, n0, 0),  acc);
      acc = wmma_bf16(a_frag(sB, 64, 32),  b_frag(wDecIh, 64, n0, 32), acc);
      acc = wmma_bf16(a_frag(hbf, 64, 0),  b_frag(wDecHh, 64, n0, 0),  acc);
      acc = wmma_bf16(a_frag(hbf, 64, 32), b_frag(wDecHh, 64, n0, 32), acc);
      scatter_d(sG, 256, n0, acc);
    }
    __syncthreads();

    for (int i = tid; i < BT * 64; i += NTH) {
      int m = i >> 6, h = i & 63;
      const float* g = &sG[m * 256];
      float ig = fast_sigmoid(g[h]);
      float fg = fast_sigmoid(g[64 + h]);
      float gg = fast_tanh(g[128 + h]);
      float og = fast_sigmoid(g[192 + h]);
      float cn = fg * cf[i] + ig * gg;
      float hn = og * fast_tanh(cn);
      cf[i] = cn; hf[i] = hn; hbf[i] = (bf16_t)hn;
    }
    __syncthreads();

    // pred = h @ outW^T + b ; output = pred + seq_last ; feed back
    if (tid < BT * CDIM) {
      int m = tid >> 3, c = tid & 7;
      float s = bOut[c];
      const float* wr = &wOut[c * 64];
      const float* hr = &hf[m * 64];
      for (int j2 = 0; j2 < 64; ++j2) s += hr[j2] * wr[j2];
      sP[tid] = s;
      out[((size_t)(b0 + m) * TDIM + t) * CDIM + c] = s + sSeq[tid];
    }
    __syncthreads();
  }
}

// ---------------- workspace layout (bytes) ----------------
#define OFF_ENCIH 0u
#define OFF_ENCHH (OFF_ENCIH + 256u * 32u * 2u)
#define OFF_EMB   (OFF_ENCHH + 256u * 64u * 2u)
#define OFF_ATTN  (OFF_EMB   +  64u * 32u * 2u)
#define OFF_COMB  (OFF_ATTN  +  96u * 128u * 2u)
#define OFF_DECIH (OFF_COMB  +  64u * 128u * 2u)
#define OFF_DECHH (OFF_DECIH + 256u * 64u * 2u)
#define OFF_BENC  (OFF_DECHH + 256u * 64u * 2u)
#define OFF_BDEC  (OFF_BENC  + 256u * 4u)

extern "C" void kernel_launch(void* const* d_in, const int* in_sizes, int n_in,
                              void* d_out, int out_size, void* d_ws, size_t ws_size,
                              hipStream_t stream) {
  (void)in_sizes; (void)n_in; (void)out_size; (void)ws_size;
  const float* x_enc   = (const float*)d_in[0];
  const float* enc_Wih = (const float*)d_in[4];
  const float* enc_Whh = (const float*)d_in[5];
  const float* enc_bih = (const float*)d_in[6];
  const float* enc_bhh = (const float*)d_in[7];
  const float* emb_W   = (const float*)d_in[8];
  const float* emb_b   = (const float*)d_in[9];
  const float* attn_W  = (const float*)d_in[10];
  const float* attn_b  = (const float*)d_in[11];
  const float* comb_W  = (const float*)d_in[12];
  const float* comb_b  = (const float*)d_in[13];
  const float* dec_Wih = (const float*)d_in[14];
  const float* dec_Whh = (const float*)d_in[15];
  const float* dec_bih = (const float*)d_in[16];
  const float* dec_bhh = (const float*)d_in[17];
  const float* out_W   = (const float*)d_in[18];
  const float* out_b   = (const float*)d_in[19];

  char* ws = (char*)d_ws;
  bf16_t* wEncIh = (bf16_t*)(ws + OFF_ENCIH);
  bf16_t* wEncHh = (bf16_t*)(ws + OFF_ENCHH);
  bf16_t* wEmb   = (bf16_t*)(ws + OFF_EMB);
  bf16_t* wAttn  = (bf16_t*)(ws + OFF_ATTN);
  bf16_t* wComb  = (bf16_t*)(ws + OFF_COMB);
  bf16_t* wDecIh = (bf16_t*)(ws + OFF_DECIH);
  bf16_t* wDecHh = (bf16_t*)(ws + OFF_DECHH);
  float*  bEnc   = (float*)(ws + OFF_BENC);
  float*  bDec   = (float*)(ws + OFF_BDEC);

  const int TPB = 256;
  auto g = [&](int n) { return dim3((unsigned)((n + TPB - 1) / TPB)); };
  cvt_pad_kernel<<<g(256 * 32), TPB, 0, stream>>>(enc_Wih, wEncIh, 256, 8, 32);
  cvt_pad_kernel<<<g(256 * 64), TPB, 0, stream>>>(enc_Whh, wEncHh, 256, 64, 64);
  cvt_pad_kernel<<<g(64 * 32),  TPB, 0, stream>>>(emb_W,   wEmb,   64, 8, 32);
  cvt_pad_kernel<<<g(96 * 128), TPB, 0, stream>>>(attn_W,  wAttn,  96, 128, 128);
  cvt_pad_kernel<<<g(64 * 128), TPB, 0, stream>>>(comb_W,  wComb,  64, 128, 128);
  cvt_pad_kernel<<<g(256 * 64), TPB, 0, stream>>>(dec_Wih, wDecIh, 256, 64, 64);
  cvt_pad_kernel<<<g(256 * 64), TPB, 0, stream>>>(dec_Whh, wDecHh, 256, 64, 64);
  bias_add_kernel<<<1, 256, 0, stream>>>(enc_bih, enc_bhh, bEnc, 256);
  bias_add_kernel<<<1, 256, 0, stream>>>(dec_bih, dec_bhh, bDec, 256);

  const size_t dynLds = (size_t)LDIM * BT * HDIM * sizeof(bf16_t); // 192KB enc_out tile
  attn_lstm_main<<<dim3(BDIM / BT), dim3(NTH), dynLds, stream>>>(
      x_enc, wEncIh, wEncHh, bEnc, wEmb, emb_b, wAttn, attn_b, wComb, comb_b,
      wDecIh, wDecHh, bDec, out_W, out_b, (float*)d_out);
}
